// GNN_41644002902305
// MI455X (gfx1250) — compile-verified, matrix operand order checked by reading, not run on hardware
//
#include <hip/hip_runtime.h>

typedef __attribute__((ext_vector_type(16))) _Float16 v16h;
typedef __attribute__((ext_vector_type(8)))  _Float16 v8h;
typedef __attribute__((ext_vector_type(8)))  float    v8f;

#define B_   8
#define N_   2048
#define D_   128
#define L_   3
#define EPSN 1e-5f

__device__ __forceinline__ float fast_sqrt(float x) {
#if __has_builtin(__builtin_amdgcn_sqrtf)
    return __builtin_amdgcn_sqrtf(x);
#else
    return sqrtf(x);
#endif
}

__device__ __forceinline__ float fast_tanh(float x) {
#if __has_builtin(__builtin_amdgcn_tanhf)
    return __builtin_amdgcn_tanhf(x);
#else
    return tanhf(x);
#endif
}

// Load a 16-wide f16 fragment as two b128 loads (K and K+16 halves).
__device__ __forceinline__ v16h load_frag(const _Float16* p) {
    v8h lo = *(const v8h*)(p);
    v8h hi = *(const v8h*)(p + 16);
    return __builtin_shufflevector(lo, hi, 0,1,2,3,4,5,6,7,8,9,10,11,12,13,14,15);
}

// ---------------- degree / dinv ----------------
__global__ __launch_bounds__(256) void k_dinv(const float* __restrict__ inst,
                                              float* __restrict__ dinv) {
    __shared__ float cx[N_], cy[N_];
    const int b = blockIdx.y, t = threadIdx.x;
    const float* ib = inst + (size_t)b * N_ * 2;
    for (int m = t; m < N_; m += 256) { cx[m] = ib[2*m]; cy[m] = ib[2*m + 1]; }
    __syncthreads();
    const int n = blockIdx.x * 256 + t;
    const float xn = cx[n], yn = cy[n];
    float deg = 1.0f;                       // self-loop weight
    for (int m = 0; m < N_; ++m) {
        float dx = xn - cx[m], dy = yn - cy[m];
        deg += fast_sqrt(dx*dx + dy*dy);    // diagonal contributes exact 0
    }
    dinv[b * N_ + n] = rsqrtf(deg);
}

// ---------------- Wg -> f16 ----------------
__global__ void k_wg16(const float* __restrict__ Wg, _Float16* __restrict__ Wg16, int n) {
    int i = blockIdx.x * 256 + threadIdx.x;
    if (i < n) Wg16[i] = (_Float16)Wg[i];
}

// ---------------- node projection ----------------
__global__ void k_proj(const float* __restrict__ inst, const float* __restrict__ Wn,
                       const float* __restrict__ bn,
                       float* __restrict__ hF, _Float16* __restrict__ h16) {
    int i  = blockIdx.x * 256 + threadIdx.x;   // flat over B*N*D
    int d  = i & (D_ - 1);
    int bn_ = i >> 7;                          // b*N + n
    float x = inst[2*bn_], y = inst[2*bn_ + 1];
    float v = x * Wn[2*d] + y * Wn[2*d + 1] + bn[d];
    hF[i] = v;
    h16[i] = (_Float16)v;
}

// ---------------- X^T = (h @ Wg[l]^T)^T  (WMMA; output stored [b][e][n]) ----------------
__global__ __launch_bounds__(256) void k_hw(const _Float16* __restrict__ h16,
                                            const _Float16* __restrict__ Wgl,
                                            _Float16* __restrict__ X16T,
                                            float* __restrict__ gSum,
                                            float* __restrict__ gSq) {
    __shared__ __align__(16) _Float16 sW[D_][D_];   // Wg[l] row-major: [e][d], K-contiguous
    const int t = threadIdx.x, b = blockIdx.y, n0 = blockIdx.x * 128;

    // zero GraphNorm stats for the upcoming k_gemm (one block is enough)
    if (blockIdx.x == 0 && b == 0) {
        for (int i = t; i < B_ * D_; i += 256) { gSum[i] = 0.f; gSq[i] = 0.f; }
    }
    for (int i = t; i < (D_ * D_) / 8; i += 256)
        ((v8h*)&sW[0][0])[i] = ((const v8h*)Wgl)[i];
    __syncthreads();

    const int lane = t & 31, w = t >> 5;            // 8 waves, each 16 rows x 128 cols
    const int col16 = lane & 15;
    const int kb   = (lane < 16) ? 0 : 8;
    const int roff = (lane < 16) ? 0 : 8;
    const int row  = n0 + w * 16 + col16;           // A-fragment row for this lane

    v8f zero; for (int i = 0; i < 8; ++i) zero[i] = 0.f;
    v8f acc[8]; for (int j = 0; j < 8; ++j) acc[j] = zero;

    for (int kc = 0; kc < D_ / 32; ++kc) {
        v16h a = load_frag(h16 + ((size_t)(b * N_ + row) * D_ + kc * 32 + kb));
        v16h bf[8];
        for (int j = 0; j < 8; ++j)
            bf[j] = load_frag(&sW[j * 16 + col16][kc * 32 + kb]);
        for (int j = 0; j < 8; ++j)
            acc[j] = __builtin_amdgcn_wmma_f32_16x16x32_f16(
                false, a, false, bf[j], (short)0, acc[j], false, false);
    }
    // store transposed: rows of acc are consecutive n -> one b128 store per fragment
    for (int j = 0; j < 8; ++j) {
        int e = j * 16 + col16;
        v8h out;
        for (int r = 0; r < 8; ++r) out[r] = (_Float16)acc[j][r];
        *(v8h*)(X16T + (size_t)(b * D_ + e) * N_ + n0 + w * 16 + roff) = out;
    }
}

// ---------------- Z = An @ X fused: An on the fly + bias/tanh/residual/stats ----------------
__global__ __launch_bounds__(128) void k_gemm(const float* __restrict__ inst,
                                              const float* __restrict__ dinv,
                                              const _Float16* __restrict__ X16T,
                                              const float* __restrict__ bgl,
                                              float* __restrict__ hF,
                                              float* __restrict__ gSum,
                                              float* __restrict__ gSq) {
    __shared__ __align__(16) _Float16 sA[2][64][32];  // double-buffered An tile (M x K)
    __shared__ float rX[64], rY[64], rDi[64];
    __shared__ float cX[2][32], cY[2][32], cDi[2][32];
    __shared__ float sSum[D_], sSq[D_];

    const int t = threadIdx.x, b = blockIdx.y, n0 = blockIdx.x * 64;
    const float* ib = inst + (size_t)b * N_ * 2;
    const float* db = dinv + b * N_;

    if (t < 64) { rX[t] = ib[2*(n0+t)]; rY[t] = ib[2*(n0+t)+1]; rDi[t] = db[n0+t]; }
    if (t < D_) { sSum[t] = 0.f; sSq[t] = 0.f; }
    if (t < 32) { cX[0][t] = ib[2*t]; cY[0][t] = ib[2*t+1]; cDi[0][t] = db[t]; }

    const int lane = t & 31, w = t >> 5;             // 4 waves, each 16 rows x 128 cols
    const int col16 = lane & 15;
    const int kb   = (lane < 16) ? 0 : 8;
    const int roff = (lane < 16) ? 0 : 8;
    const int arow = t >> 1;                          // An compute: row 0..63
    const int ak0  = (t & 1) * 16;                    // 16 K-values per thread

    v8f zero; for (int i = 0; i < 8; ++i) zero[i] = 0.f;
    v8f acc[8]; for (int j = 0; j < 8; ++j) acc[j] = zero;
    __syncthreads();

    const float xr = rX[arow], yr = rY[arow], dr = rDi[arow];
    const int   gn = n0 + arow;

    for (int kc = 0; kc < N_ / 32; ++kc) {
        const int p  = kc & 1;
        const int m0 = kc * 32;
        // compute An tile (normalized distances) into sA[p] as f16
        {
            v8h tlo, thi;
            for (int kk = 0; kk < 16; ++kk) {
                int k = ak0 + kk, gm = m0 + k;
                float dx = xr - cX[p][k], dy = yr - cY[p][k];
                float v = fast_sqrt(dx*dx + dy*dy);
                if (gn == gm) v += 1.0f;               // self-loop
                _Float16 h = (_Float16)(v * dr * cDi[p][k]);
                if (kk < 8) tlo[kk] = h; else thi[kk - 8] = h;
            }
            *(v8h*)&sA[p][arow][ak0]     = tlo;
            *(v8h*)&sA[p][arow][ak0 + 8] = thi;
        }
        // prefetch next chunk's column coords into the other buffer
        if (kc + 1 < N_ / 32 && t < 32) {
            int m1 = m0 + 32;
            cX[p ^ 1][t] = ib[2*(m1+t)];
            cY[p ^ 1][t] = ib[2*(m1+t)+1];
            cDi[p ^ 1][t] = db[m1+t];
        }
        __syncthreads();   // single barrier per chunk (double-buffered WAR-safe)
        // WMMA: A from LDS, B fragments direct from global X16T (K-contiguous rows)
        {
            v16h a = load_frag(&sA[p][w * 16 + col16][0] + kb);
            v16h bf[8];
            for (int j = 0; j < 8; ++j) {
                int d = j * 16 + col16;
                bf[j] = load_frag(X16T + (size_t)(b * D_ + d) * N_ + m0 + kb);
            }
            for (int j = 0; j < 8; ++j)
                acc[j] = __builtin_amdgcn_wmma_f32_16x16x32_f16(
                    false, a, false, bf[j], (short)0, acc[j], false, false);
        }
    }

    // epilogue: bias + tanh + residual, accumulate GraphNorm stats
    for (int j = 0; j < 8; ++j) {
        int d = j * 16 + col16;
        float bgv = bgl[d];
        float s = 0.f, ss = 0.f;
        for (int r = 0; r < 8; ++r) {
            int n = n0 + w * 16 + r + roff;
            size_t idx = (size_t)(b * N_ + n) * D_ + d;
            float hp = fast_tanh(acc[j][r] + bgv) + hF[idx];
            hF[idx] = hp;
            s += hp; ss += hp * hp;
        }
        atomicAdd(&sSum[d], s);
        atomicAdd(&sSq[d], ss);
    }
    __syncthreads();
    if (t < D_) {
        atomicAdd(&gSum[b * D_ + t], sSum[t]);
        atomicAdd(&gSq[b * D_ + t],  sSq[t]);
    }
}

// ---------------- GraphNorm ----------------
__global__ __launch_bounds__(256) void k_norm(const float* __restrict__ hIn,
                                              const float* __restrict__ gSum,
                                              const float* __restrict__ gSq,
                                              const float* __restrict__ gw,
                                              const float* __restrict__ gb,
                                              const float* __restrict__ ga,
                                              float* __restrict__ outF,
                                              _Float16* __restrict__ h16,
                                              int writeH16) {
    const int t = threadIdx.x, b = blockIdx.y, n0 = blockIdx.x * 16;
    const int d = t & (D_ - 1);
    const int r = t >> 7;                       // 0..1
    float a    = ga[d];
    float mean = gSum[b * D_ + d] * (1.0f / N_);
    float var  = gSq[b * D_ + d] * (1.0f / N_) - a * (2.0f - a) * mean * mean;
    float rs   = rsqrtf(var + EPSN);
    float wv = gw[d], bv = gb[d], am = a * mean;
    for (int i = 0; i < 8; ++i) {
        int n = n0 + r + 2 * i;
        size_t idx = (size_t)(b * N_ + n) * D_ + d;
        float y = wv * (hIn[idx] - am) * rs + bv;
        outF[idx] = y;
        if (writeH16) h16[idx] = (_Float16)y;
    }
}

extern "C" void kernel_launch(void* const* d_in, const int* in_sizes, int n_in,
                              void* d_out, int out_size, void* d_ws, size_t ws_size,
                              hipStream_t stream) {
    (void)in_sizes; (void)n_in; (void)out_size; (void)ws_size;
    const float* inst = (const float*)d_in[0];
    const float* Wn   = (const float*)d_in[1];
    const float* bn   = (const float*)d_in[2];
    const float* Wg   = (const float*)d_in[3];
    const float* bg   = (const float*)d_in[4];
    const float* gw   = (const float*)d_in[5];
    const float* gb   = (const float*)d_in[6];
    const float* ga   = (const float*)d_in[7];

    char* w = (char*)d_ws;
    size_t off = 0;
    auto alloc = [&](size_t bytes) {
        void* p = w + off;
        off += (bytes + 255) & ~(size_t)255;
        return p;
    };
    float*    hF    = (float*)   alloc((size_t)B_ * N_ * D_ * 4);
    _Float16* h16   = (_Float16*)alloc((size_t)B_ * N_ * D_ * 2);
    _Float16* X16T  = (_Float16*)alloc((size_t)B_ * N_ * D_ * 2);
    _Float16* Wg16  = (_Float16*)alloc((size_t)L_ * D_ * D_ * 2);
    float*    dinv  = (float*)   alloc((size_t)B_ * N_ * 4);
    float*    gS    = (float*)   alloc((size_t)B_ * D_ * 4);
    float*    gQ    = (float*)   alloc((size_t)B_ * D_ * 4);

    k_dinv<<<dim3(N_ / 256, B_), 256, 0, stream>>>(inst, dinv);
    k_wg16<<<(L_ * D_ * D_ + 255) / 256, 256, 0, stream>>>(Wg, Wg16, L_ * D_ * D_);
    k_proj<<<(B_ * N_ * D_) / 256, 256, 0, stream>>>(inst, Wn, bn, hF, h16);

    for (int l = 0; l < L_; ++l) {
        k_hw<<<dim3(N_ / 128, B_), 256, 0, stream>>>(h16, Wg16 + (size_t)l * D_ * D_,
                                                     X16T, gS, gQ);
        k_gemm<<<dim3(N_ / 64, B_), 128, 0, stream>>>(inst, dinv, X16T, bg + l * D_,
                                                      hF, gS, gQ);
        int last = (l == L_ - 1);
        k_norm<<<dim3(N_ / 16, B_), 256, 0, stream>>>(hF, gS, gQ, gw, gb, ga,
                                                      last ? (float*)d_out : hF,
                                                      h16, last ? 0 : 1);
    }
}
